// ConvLayer_50714973831732
// MI455X (gfx1250) — matrix operand order repaired and checked by reading, not compiled, not run on hardware
//
#include <hip/hip_runtime.h>
#include <hip/hip_bf16.h>

// ---------------------------------------------------------------------------
// CDNA5 (gfx1250) WMMA GNN conv layer.
//   pass 0: zero agg, convert x -> bf16 once (saves ~64x redundant cvt + 2x L2 bytes)
//   pass 1: edge MLP (2 types, 64->64->64, bf16 WMMA, f32 acc) -> atomic scatter
//   pass 2: node MLP (96->64->64->32, bf16 WMMA, f32 acc) + skip connection
// Transposed formulation: A = W^T (staged bf16 in LDS), B = activations^T
// so edges/nodes ride the N dimension (per-lane scalar edge_prob / scatter).
// ---------------------------------------------------------------------------

typedef __attribute__((ext_vector_type(16))) __bf16 v16bf;
typedef __attribute__((ext_vector_type(8)))  __bf16 v8bf;
typedef __attribute__((ext_vector_type(8)))  float  v8f;
typedef __attribute__((ext_vector_type(4)))  float  v4f;

__device__ __forceinline__ float relu1(float v) {
  // maps to a single v_max_num_f32 (NaN -> 0, same as max_num(x, 0))
  return v > 0.0f ? v : 0.0f;
}

__device__ __forceinline__ v8f wmma_bf16(v16bf a, v16bf b, v8f c) {
  // v_wmma_f32_16x16x32_bf16 (neg_a, A, neg_b, B, c_mod, C, reuse_a, reuse_b)
  return __builtin_amdgcn_wmma_f32_16x16x32_bf16(false, a, false, b, (short)0, c,
                                                 false, false);
}

// A-matrix fragment (16x32 bf16) for one K-chunk of a row-contiguous W^T row.
// Lane (h = lane>>4, m = lane&15): elems 0..15 -> K = kbase+16h+j.
__device__ __forceinline__ v16bf load_afrag(const __bf16* wrow, int kbase, int h) {
  union { v16bf v; v8bf p[2]; } u;
  u.p[0] = *(const v8bf*)(wrow + kbase + 16 * h);
  u.p[1] = *(const v8bf*)(wrow + kbase + 16 * h + 8);
  return u.v;
}

// B-matrix fragment: 16 contiguous bf16 (16B aligned), lane (h,n) K = kbase+16h+j.
__device__ __forceinline__ v16bf load_bfrag(const __bf16* p) {
  union { v16bf v; v8bf q[2]; } u;
  u.q[0] = ((const v8bf*)p)[0];
  u.q[1] = ((const v8bf*)p)[1];
  return u.v;
}

// 16 contiguous f32 from global -> v16bf (for agg, which must stay f32).
__device__ __forceinline__ v16bf cvt16(const float* p) {
  union { v16bf v; __bf16 e[16]; } u;
#pragma unroll
  for (int c = 0; c < 4; ++c) {
    v4f f = ((const v4f*)p)[c];
#pragma unroll
    for (int j = 0; j < 4; ++j) u.e[4 * c + j] = (__bf16)f[j];
  }
  return u.v;
}

// 8 contiguous f32 (bias broadcast along N: C elem r -> M = r + 8h).
__device__ __forceinline__ v8f load_bias8(const float* b) {
  union { v8f v; v4f q[2]; } u;
  u.q[0] = ((const v4f*)b)[0];
  u.q[1] = ((const v4f*)b)[1];
  return u.v;
}

// relu + f32->bf16 pack + single 16B LDS store.
__device__ __forceinline__ void store8_relu(__bf16* dst, v8f v) {
  union { v8bf b; __bf16 e[8]; } u;
#pragma unroll
  for (int j = 0; j < 8; ++j) u.e[j] = (__bf16)relu1(v[j]);
  *(v8bf*)dst = u.b;
}

// ---------------------------------------------------------------------------
__global__ void zero_f32_kernel(float* __restrict__ p, int count) {
  int i = blockIdx.x * blockDim.x + threadIdx.x;
  if (i < count) p[i] = 0.0f;
}

// Convert x (f32) to bf16 copy, 8 elements per thread (vectorized b128 in/b128 out).
__global__ void cvt_x_bf16_kernel(const float* __restrict__ x,
                                  __bf16* __restrict__ xb, int count8) {
  int i = blockIdx.x * blockDim.x + threadIdx.x;
  if (i < count8) {
    const float* p = x + (long long)i * 8;
    union { v8bf b; __bf16 e[8]; } u;
#pragma unroll
    for (int c = 0; c < 2; ++c) {
      v4f f = ((const v4f*)p)[c];
#pragma unroll
      for (int j = 0; j < 4; ++j) u.e[4 * c + j] = (__bf16)f[j];
    }
    *(v8bf*)(xb + (long long)i * 8) = u.b;
  }
}

// ---------------------------------------------------------------------------
struct __align__(32) EdgeSmem {
  __bf16 w1t[2][64 * 64];   // W1[t]^T : [out][in] bf16
  __bf16 w2t[2][64 * 64];   // W2[t]^T
  float  b1s[2][64];
  float  b2s[2][64];
  __bf16 scr[8][16][64];    // per-wave layer1->layer2 transpose scratch
};

__global__ __launch_bounds__(256) void edge_msg_kernel(
    const __bf16* __restrict__ xb, const float* __restrict__ eprob,
    const float* __restrict__ W1, const float* __restrict__ B1,
    const float* __restrict__ W2, const float* __restrict__ B2,
    const int* __restrict__ eidx, float* __restrict__ agg,
    int E, int ntiles) {
  __shared__ EdgeSmem sm;

  // Stage bf16-transposed weights + biases once per block.
  for (int idx = threadIdx.x; idx < 2 * 64 * 64; idx += blockDim.x) {
    int t = idx >> 12, rem = idx & 4095, i = rem >> 6, o = rem & 63;
    sm.w1t[t][o * 64 + i] = (__bf16)W1[idx];
    sm.w2t[t][o * 64 + i] = (__bf16)W2[idx];
  }
  for (int idx = threadIdx.x; idx < 2 * 64; idx += blockDim.x) {
    sm.b1s[idx >> 6][idx & 63] = B1[idx];
    sm.b2s[idx >> 6][idx & 63] = B2[idx];
  }
  __syncthreads();

  const int lane = threadIdx.x & 31;
  const int wv   = threadIdx.x >> 5;
  const int h    = lane >> 4;    // K-half selector
  const int n    = lane & 15;    // N column (= edge within tile) / A row
  const int* rowp = eidx;
  const int* colp = eidx + E;
  __bf16* myscr = &sm.scr[wv][n][0];

  const int wgid = blockIdx.x * (blockDim.x >> 5) + wv;
  const int wcnt = gridDim.x * (blockDim.x >> 5);

  for (int tile = wgid; tile < ntiles; tile += wcnt) {
    const int  e     = tile * 16 + n;
    const bool valid = (e < E);
    const int  ec    = valid ? e : 0;
    const int  ridx  = rowp[ec];
    const int  cidx  = colp[ec];

    // B = pre_msg^T (64 x 16): K-chunk0 = xb[row], K-chunk1 = xb[col]
    v16bf bA = load_bfrag(xb + (long long)ridx * 32 + 16 * h);
    v16bf bB = load_bfrag(xb + (long long)cidx * 32 + 16 * h);
    float ep0 = valid ? eprob[ec] : 0.0f;
    float ep1 = valid ? eprob[E + ec] : 0.0f;
    float epv[2] = {ep0, ep1};

    v8f msg[4];
#pragma unroll
    for (int mt = 0; mt < 4; ++mt)
#pragma unroll
      for (int r = 0; r < 8; ++r) msg[mt][r] = 0.0f;

#pragma unroll
    for (int t = 0; t < 2; ++t) {
      // Layer 1: m1^T = relu(W1^T @ pre_msg^T + b1)  -> scr (bf16)
#pragma unroll
      for (int mt = 0; mt < 4; ++mt) {
        v8f acc = load_bias8(&sm.b1s[t][mt * 16 + 8 * h]);
        const __bf16* wr = &sm.w1t[t][(mt * 16 + n) * 64];
        acc = wmma_bf16(load_afrag(wr, 0, h), bA, acc);
        acc = wmma_bf16(load_afrag(wr, 32, h), bB, acc);
        store8_relu(myscr + mt * 16 + 8 * h, acc);
      }
      // Reload m1^T as B fragments (same-wave LDS ops are in-order).
      v16bf m0 = load_bfrag(myscr + 16 * h);
      v16bf m1 = load_bfrag(myscr + 32 + 16 * h);
      // Layer 2: msg += relu(W2^T @ m1^T + b2) * edge_prob[t]
#pragma unroll
      for (int mt = 0; mt < 4; ++mt) {
        v8f acc = load_bias8(&sm.b2s[t][mt * 16 + 8 * h]);
        const __bf16* wr = &sm.w2t[t][(mt * 16 + n) * 64];
        acc = wmma_bf16(load_afrag(wr, 0, h), m0, acc);
        acc = wmma_bf16(load_afrag(wr, 32, h), m1, acc);
#pragma unroll
        for (int r = 0; r < 8; ++r)
          msg[mt][r] += relu1(acc[r]) * epv[t];
      }
    }

    // Scatter-add: lane owns edge (base+n), features mt*16 + 8h + r.
    // Invalid lanes contribute exact 0.0f (harmless atomic).
    float* arow = agg + (long long)ridx * 64 + 8 * h;
#pragma unroll
    for (int mt = 0; mt < 4; ++mt)
#pragma unroll
      for (int r = 0; r < 8; ++r)
        atomicAdd(arow + mt * 16 + r, msg[mt][r]);
  }
}

// ---------------------------------------------------------------------------
struct __align__(32) NodeSmem {
  __bf16 wo1t[64 * 96];    // Wo1^T : [64 out][96 in]
  __bf16 wo2t[64 * 64];
  __bf16 wo3t[32 * 64];
  float  bo1s[64];
  float  bo2s[64];
  float  bo3s[32];
  __bf16 scrA[8][16][64];
  __bf16 scrB[8][16][64];
};

__global__ __launch_bounds__(256) void node_out_kernel(
    const float* __restrict__ x, const __bf16* __restrict__ xb,
    const float* __restrict__ agg,
    const float* __restrict__ Wo1, const float* __restrict__ bo1,
    const float* __restrict__ Wo2, const float* __restrict__ bo2,
    const float* __restrict__ Wo3, const float* __restrict__ bo3,
    float* __restrict__ out, int NN, int ntiles) {
  __shared__ NodeSmem sm;

  for (int idx = threadIdx.x; idx < 96 * 64; idx += blockDim.x) {
    int i = idx >> 6, o = idx & 63;
    sm.wo1t[o * 96 + i] = (__bf16)Wo1[idx];
  }
  for (int idx = threadIdx.x; idx < 64 * 64; idx += blockDim.x) {
    int i = idx >> 6, o = idx & 63;
    sm.wo2t[o * 64 + i] = (__bf16)Wo2[idx];
  }
  for (int idx = threadIdx.x; idx < 64 * 32; idx += blockDim.x) {
    int i = idx >> 5, o = idx & 31;
    sm.wo3t[o * 64 + i] = (__bf16)Wo3[idx];
  }
  for (int idx = threadIdx.x; idx < 64; idx += blockDim.x) {
    sm.bo1s[idx] = bo1[idx];
    sm.bo2s[idx] = bo2[idx];
    if (idx < 32) sm.bo3s[idx] = bo3[idx];
  }
  __syncthreads();

  const int lane = threadIdx.x & 31;
  const int wv   = threadIdx.x >> 5;
  const int h    = lane >> 4;
  const int n    = lane & 15;

  const int wgid = blockIdx.x * (blockDim.x >> 5) + wv;
  const int wcnt = gridDim.x * (blockDim.x >> 5);

  for (int tile = wgid; tile < ntiles; tile += wcnt) {
    const int  node  = tile * 16 + n;
    const bool valid = (node < NN);
    const int  nc    = valid ? node : 0;

    // B = aug^T (96 x 16): kc0 = x (bf16 copy), kc1/kc2 = agg (f32 -> cvt)
    v16bf bx = load_bfrag(xb + (long long)nc * 32 + 16 * h);
    v16bf g0 = cvt16(agg + (long long)nc * 64 + 16 * h);
    v16bf g1 = cvt16(agg + (long long)nc * 64 + 32 + 16 * h);

    // Layer 1: relu(Wo1^T @ aug^T + bo1) -> scrA
#pragma unroll
    for (int mt = 0; mt < 4; ++mt) {
      v8f acc = load_bias8(&sm.bo1s[mt * 16 + 8 * h]);
      const __bf16* wr = &sm.wo1t[(mt * 16 + n) * 96];
      acc = wmma_bf16(load_afrag(wr, 0, h), bx, acc);
      acc = wmma_bf16(load_afrag(wr, 32, h), g0, acc);
      acc = wmma_bf16(load_afrag(wr, 64, h), g1, acc);
      store8_relu(&sm.scrA[wv][n][mt * 16 + 8 * h], acc);
    }
    v16bf p0 = load_bfrag(&sm.scrA[wv][n][16 * h]);
    v16bf p1 = load_bfrag(&sm.scrA[wv][n][32 + 16 * h]);

    // Layer 2: relu(Wo2^T @ p + bo2) -> scrB (double buffer vs scrA)
#pragma unroll
    for (int mt = 0; mt < 4; ++mt) {
      v8f acc = load_bias8(&sm.bo2s[mt * 16 + 8 * h]);
      const __bf16* wr = &sm.wo2t[(mt * 16 + n) * 64];
      acc = wmma_bf16(load_afrag(wr, 0, h), p0, acc);
      acc = wmma_bf16(load_afrag(wr, 32, h), p1, acc);
      store8_relu(&sm.scrB[wv][n][mt * 16 + 8 * h], acc);
    }
    v16bf q0 = load_bfrag(&sm.scrB[wv][n][16 * h]);
    v16bf q1 = load_bfrag(&sm.scrB[wv][n][32 + 16 * h]);

    // Layer 3: Wo3^T @ q + bo3, + skip(x f32), store 32 feats.
#pragma unroll
    for (int mt = 0; mt < 2; ++mt) {
      v8f acc = load_bias8(&sm.bo3s[mt * 16 + 8 * h]);
      const __bf16* wr = &sm.wo3t[(mt * 16 + n) * 64];
      acc = wmma_bf16(load_afrag(wr, 0, h), q0, acc);
      acc = wmma_bf16(load_afrag(wr, 32, h), q1, acc);
      if (valid) {
        const float* xr = x + (long long)node * 32 + mt * 16 + 8 * h;
        float* orow = out + (long long)node * 32 + mt * 16 + 8 * h;
#pragma unroll
        for (int r = 0; r < 8; ++r) orow[r] = acc[r] + xr[r];
      }
    }
  }
}

// ---------------------------------------------------------------------------
extern "C" void kernel_launch(void* const* d_in, const int* in_sizes, int n_in,
                              void* d_out, int out_size, void* d_ws, size_t ws_size,
                              hipStream_t stream) {
  const float* x    = (const float*)d_in[0];
  const float* ep   = (const float*)d_in[1];
  const float* W1   = (const float*)d_in[2];
  const float* B1   = (const float*)d_in[3];
  const float* W2   = (const float*)d_in[4];
  const float* B2   = (const float*)d_in[5];
  const float* Wo1  = (const float*)d_in[6];
  const float* bo1  = (const float*)d_in[7];
  const float* Wo2  = (const float*)d_in[8];
  const float* bo2  = (const float*)d_in[9];
  const float* Wo3  = (const float*)d_in[10];
  const float* bo3  = (const float*)d_in[11];
  const int*   eidx = (const int*)d_in[12];

  const int E  = in_sizes[1] / 2;    // edge_prob is [2, E]
  const int NN = in_sizes[0] / 32;   // x is [1, N, 32]

  // workspace layout: agg [N,64] f32, then xb [N,32] bf16
  float*  agg = (float*)d_ws;
  __bf16* xb  = (__bf16*)((char*)d_ws + (size_t)NN * 64 * sizeof(float));
  float*  out = (float*)d_out;

  // agg must be zero every call (workspace is not re-poisoned between replays).
  const int aggn = NN * 64;
  zero_f32_kernel<<<(aggn + 255) / 256, 256, 0, stream>>>(agg, aggn);

  // one-time (per call) f32 -> bf16 copy of x
  const int c8 = NN * 32 / 8;
  cvt_x_bf16_kernel<<<(c8 + 255) / 256, 256, 0, stream>>>(x, xb, c8);

  const int etiles  = (E + 15) / 16;
  int eblocks = (etiles + 7) / 8;
  if (eblocks > 2048) eblocks = 2048;
  edge_msg_kernel<<<eblocks, 256, 0, stream>>>(xb, ep, W1, B1, W2, B2, eidx, agg,
                                               E, etiles);

  const int ntls = (NN + 15) / 16;
  int nblocks = (ntls + 7) / 8;
  if (nblocks > 2048) nblocks = 2048;
  node_out_kernel<<<nblocks, 256, 0, stream>>>(x, xb, agg, Wo1, bo1, Wo2, bo2,
                                               Wo3, bo3, out, NN, ntls);
}
